// Attention_23287312678912
// MI455X (gfx1250) — matrix operand order repaired and checked by reading, not compiled, not run on hardware
//
#include <hip/hip_runtime.h>

typedef __attribute__((ext_vector_type(16))) __bf16 v16bf;
typedef __attribute__((ext_vector_type(8)))  __bf16 v8bf;
typedef __attribute__((ext_vector_type(8)))  float  v8f;

#if __has_builtin(__builtin_amdgcn_exp2f)
#define EXP2F(x) __builtin_amdgcn_exp2f(x)
#else
#define EXP2F(x) exp2f(x)
#endif

#define SEQ 2048
#define DK  64
#define NBH 32          // B*h = 2*16
#define QBLK 128        // query rows per workgroup (8 waves x 16)
#define KBLK 32         // key rows per LDS stage

// softmax runs in log2 domain: fold 1/sqrt(64) * log2(e) into Q once
#define QSCALE (0.125f * 1.44269504088896340736f)
#define NEG_BIG (-1e30f)

__device__ __forceinline__ v16bf pack16(float4 a, float4 b, float4 c, float4 d, float s) {
    v16bf r;
    r[0]=(__bf16)(a.x*s);  r[1]=(__bf16)(a.y*s);  r[2]=(__bf16)(a.z*s);  r[3]=(__bf16)(a.w*s);
    r[4]=(__bf16)(b.x*s);  r[5]=(__bf16)(b.y*s);  r[6]=(__bf16)(b.z*s);  r[7]=(__bf16)(b.w*s);
    r[8]=(__bf16)(c.x*s);  r[9]=(__bf16)(c.y*s);  r[10]=(__bf16)(c.z*s); r[11]=(__bf16)(c.w*s);
    r[12]=(__bf16)(d.x*s); r[13]=(__bf16)(d.y*s); r[14]=(__bf16)(d.z*s); r[15]=(__bf16)(d.w*s);
    return r;
}

__device__ __forceinline__ float rowmax16(float v) {
    v = fmaxf(v, __shfl_xor(v, 1, 32));
    v = fmaxf(v, __shfl_xor(v, 2, 32));
    v = fmaxf(v, __shfl_xor(v, 4, 32));
    v = fmaxf(v, __shfl_xor(v, 8, 32));
    return v;
}

__global__ __launch_bounds__(256, 1)
void fa_causal_kernel(const float* __restrict__ Qg, const float* __restrict__ Kg,
                      const float* __restrict__ Vg, float* __restrict__ Og)
{
    // Double-buffered K (row-major, bf16) and V^T (d-major, bf16); per-wave bf16 P scratch.
    __shared__ __align__(32) __bf16 sK [2][KBLK * DK];   // [key][d]
    __shared__ __align__(32) __bf16 sVT[2][DK * KBLK];   // [d][key]
    __shared__ __align__(32) __bf16 sP [8][16 * 32];     // per-wave 16x32 P tile (bf16)

    const int tid   = threadIdx.x;
    const int wave  = tid >> 5;
    const int lane  = tid & 31;
    const int l16   = lane & 15;
    const int lhalf = lane >> 4;              // 0: lanes 0-15, 1: lanes 16-31

    const int bh    = blockIdx.y;             // 0..31
    const int qwg   = blockIdx.x * QBLK;
    const int qbase = qwg + wave * 16;

    const float* Qb = Qg + (size_t)bh * SEQ * DK;
    const float* Kb = Kg + (size_t)bh * SEQ * DK;
    const float* Vb = Vg + (size_t)bh * SEQ * DK;
    float*       Ob = Og + (size_t)bh * SEQ * DK;

    // ---- Load Q tile once (scaled into log2-softmax domain), two A-fragments ----
    const float* qrow = Qb + (size_t)(qbase + l16) * DK;
    v16bf Aq[2];
    #pragma unroll
    for (int s = 0; s < 2; ++s) {
        const int d0 = s * 32 + (lhalf ? 16 : 0);
        float4 f0 = *(const float4*)(qrow + d0);
        float4 f1 = *(const float4*)(qrow + d0 + 4);
        float4 f2 = *(const float4*)(qrow + d0 + 8);
        float4 f3 = *(const float4*)(qrow + d0 + 12);
        Aq[s] = pack16(f0, f1, f2, f3, QSCALE);
    }

    // all-ones B fragment: one WMMA computes all 16 row sums of P
    v16bf Bones;
    #pragma unroll
    for (int i = 0; i < 16; ++i) Bones[i] = (__bf16)1.0f;

    // ---- Online-softmax state (log2 domain) ----
    v8f   Oacc[4] = {{}, {}, {}, {}};         // 16x64 O tile, 4 d-tiles in C-layout
    float m_run[8], l_run[8];
    #pragma unroll
    for (int r = 0; r < 8; ++r) { m_run[r] = NEG_BIG; l_run[r] = 0.0f; }

    // staging coords: each thread loads 8 floats of K and 8 of V per block
    const int srow = tid >> 3;                // 0..31 (key row)
    const int scol = (tid & 7) * 8;           // 0,8,...,56 (d col)

    const int nkb = (qwg + QBLK) / KBLK;      // causal: keys < qwg+128

    for (int kb = 0; kb < nkb; ++kb) {
        const int buf = kb & 1;

        // ---- cooperative stage of K and V^T (f32 -> bf16) ----
        const float* krow = Kb + (size_t)(kb * KBLK + srow) * DK + scol;
        const float* vrow = Vb + (size_t)(kb * KBLK + srow) * DK + scol;
        float4 k0 = *(const float4*)(krow);
        float4 k1 = *(const float4*)(krow + 4);
        float4 v0 = *(const float4*)(vrow);
        float4 v1 = *(const float4*)(vrow + 4);
        if (kb + 1 < nkb) {                   // global_prefetch_b8 for next block
            __builtin_prefetch(krow + KBLK * DK, 0, 1);
            __builtin_prefetch(vrow + KBLK * DK, 0, 1);
        }
        {   // K: row-major
            __bf16* kd = &sK[buf][srow * DK + scol];
            kd[0]=(__bf16)k0.x; kd[1]=(__bf16)k0.y; kd[2]=(__bf16)k0.z; kd[3]=(__bf16)k0.w;
            kd[4]=(__bf16)k1.x; kd[5]=(__bf16)k1.y; kd[6]=(__bf16)k1.z; kd[7]=(__bf16)k1.w;
            // V: transposed (d-major) so PV B-fragments are contiguous per lane
            __bf16* vt = &sVT[buf][0];
            vt[(scol + 0) * KBLK + srow] = (__bf16)v0.x;
            vt[(scol + 1) * KBLK + srow] = (__bf16)v0.y;
            vt[(scol + 2) * KBLK + srow] = (__bf16)v0.z;
            vt[(scol + 3) * KBLK + srow] = (__bf16)v0.w;
            vt[(scol + 4) * KBLK + srow] = (__bf16)v1.x;
            vt[(scol + 5) * KBLK + srow] = (__bf16)v1.y;
            vt[(scol + 6) * KBLK + srow] = (__bf16)v1.z;
            vt[(scol + 7) * KBLK + srow] = (__bf16)v1.w;
        }
        __syncthreads();

        // wave-uniform causal skip (EXEC stays all-1s inside)
        if (kb * KBLK <= qbase + 15) {
            // ---- scores (log2 domain): S = (Q*scale) K^T, 2 n-tiles x 2 k-slices ----
            v8f sc[2];
            #pragma unroll
            for (int j = 0; j < 2; ++j) {
                v8f c = {};
                #pragma unroll
                for (int s = 0; s < 2; ++s) {
                    const __bf16* kp = &sK[buf][(j * 16 + l16) * DK + s * 32 + (lhalf ? 16 : 0)];
                    v16bf bk = *(const v16bf*)kp;  // 32B contiguous per lane
                    c = __builtin_amdgcn_wmma_f32_16x16x32_bf16(
                            false, Aq[s], false, bk, (short)0, c, false, false);
                }
                sc[j] = c;
            }

            // ---- causal mask only on diagonal blocks (wave-uniform branch) ----
            if (kb * KBLK + (KBLK - 1) > qbase) {
                #pragma unroll
                for (int j = 0; j < 2; ++j)
                    #pragma unroll
                    for (int r = 0; r < 8; ++r) {
                        const int key = kb * KBLK + j * 16 + l16;
                        const int qi  = qbase + r + (lhalf ? 8 : 0);
                        if (key > qi) sc[j][r] = NEG_BIG;
                    }
            }

            // ---- online softmax (base-2): row max via lane shuffles ----
            float alpha[8];
            #pragma unroll
            for (int r = 0; r < 8; ++r) {
                float mx   = rowmax16(fmaxf(sc[0][r], sc[1][r]));
                float mnew = fmaxf(m_run[r], mx);
                alpha[r]   = EXP2F(m_run[r] - mnew);
                sc[0][r]   = EXP2F(sc[0][r] - mnew);
                sc[1][r]   = EXP2F(sc[1][r] - mnew);
                m_run[r]   = mnew;
            }

            // ---- P: C-layout -> A-layout via per-wave LDS bounce (bf16) ----
            __bf16* pw = sP[wave];
            #pragma unroll
            for (int r = 0; r < 8; ++r) {
                const int row = r + (lhalf ? 8 : 0);
                pw[row * 32 + l16]      = (__bf16)sc[0][r];
                pw[row * 32 + 16 + l16] = (__bf16)sc[1][r];
            }
            asm volatile("s_wait_dscnt 0" ::: "memory");  // same-wave DS in-order + hard wait
            const __bf16* pr = pw + l16 * 32 + (lhalf ? 16 : 0);
            v8bf plo = *(const v8bf*)(pr);        // K 0..7   (or 16..23)
            v8bf phi = *(const v8bf*)(pr + 8);    // K 8..15  (or 24..31)
            v16bf Ap = __builtin_shufflevector(plo, phi,
                          0,1,2,3,4,5,6,7,8,9,10,11,12,13,14,15);

            // ---- row sums via WMMA against all-ones B (replaces bpermute tree) ----
            v8f zero = {};
            v8f psum = __builtin_amdgcn_wmma_f32_16x16x32_bf16(
                           false, Ap, false, Bones, (short)0, zero, false, false);
            #pragma unroll
            for (int r = 0; r < 8; ++r)
                l_run[r] = l_run[r] * alpha[r] + psum[r];

            // ---- O = diag(alpha) * O + P V : 4 d-tiles ----
            #pragma unroll
            for (int t = 0; t < 4; ++t) {
                const __bf16* vp = &sVT[buf][(t * 16 + l16) * KBLK + (lhalf ? 16 : 0)];
                v16bf bv = *(const v16bf*)vp;
                v8f o = Oacc[t];
                #pragma unroll
                for (int r = 0; r < 8; ++r) o[r] *= alpha[r];
                Oacc[t] = __builtin_amdgcn_wmma_f32_16x16x32_bf16(
                              false, Ap, false, bv, (short)0, o, false, false);
            }
        }
    }

    // ---- epilogue: normalize and store ----
    float inv[8];
    #pragma unroll
    for (int r = 0; r < 8; ++r) inv[r] = 1.0f / l_run[r];
    #pragma unroll
    for (int t = 0; t < 4; ++t)
        #pragma unroll
        for (int r = 0; r < 8; ++r) {
            const int qi = qbase + r + (lhalf ? 8 : 0);
            Ob[(size_t)qi * DK + t * 16 + l16] = Oacc[t][r] * inv[r];
        }
}

extern "C" void kernel_launch(void* const* d_in, const int* in_sizes, int n_in,
                              void* d_out, int out_size, void* d_ws, size_t ws_size,
                              hipStream_t stream) {
    const float* Q = (const float*)d_in[0];
    const float* K = (const float*)d_in[1];
    const float* V = (const float*)d_in[2];
    float*       O = (float*)d_out;
    dim3 grid(SEQ / QBLK, NBH);   // (16, 32)
    fa_causal_kernel<<<grid, 256, 0, stream>>>(Q, K, V, O);
}
